// EdgeFPModule_47571057771114
// MI455X (gfx1250) — matrix operand order repaired and checked by reading, not compiled, not run on hardware
//
#include <hip/hip_runtime.h>

typedef __attribute__((ext_vector_type(16))) __bf16 v16bf;
typedef __attribute__((ext_vector_type(8)))  float  v8f;
typedef __attribute__((ext_vector_type(4)))  int    v4i;

#define BB    8
#define N1    4096
#define N2    1024
#define C1X   64
#define C2X   128
#define CIN1  320      // 2*C2 + C1
#define CH    256      // out channels of both convs
#define KNN   3
#define NCOLS (BB * N1 * KNN)   // 98304
#define BNEPS 1e-5f

#if __has_builtin(__builtin_amdgcn_global_load_async_to_lds_b128)
#define HAVE_ASYNC_LDS 1
#else
#define HAVE_ASYNC_LDS 0
#endif

static __device__ __forceinline__ unsigned short f2bf(float f) {
  unsigned int u = __float_as_uint(f);
  unsigned int r = (u + 0x7FFFu + ((u >> 16) & 1u)) >> 16;  // RNE
  return (unsigned short)r;
}
static __device__ __forceinline__ float bf2f(unsigned short h) {
  return __uint_as_float(((unsigned int)h) << 16);
}

// copy 16 bytes global -> LDS (async on CDNA5 if available)
static __device__ __forceinline__ void stage16(const unsigned short* gsrc,
                                               unsigned short* ldst) {
#if HAVE_ASYNC_LDS
  __builtin_amdgcn_global_load_async_to_lds_b128(
      (__attribute__((address_space(1))) v4i*)gsrc,
      (__attribute__((address_space(3))) v4i*)ldst, 0, 0);
#else
  *(uint4*)ldst = *(const uint4*)gsrc;
#endif
}

static __device__ __forceinline__ void wait_async_copies() {
#if HAVE_ASYNC_LDS
#if __has_builtin(__builtin_amdgcn_s_wait_asynccnt)
  __builtin_amdgcn_s_wait_asynccnt(0);
#else
  asm volatile("s_wait_asynccnt 0x0" ::: "memory");
#endif
#endif
}

// ---------------------------------------------------------------- KNN (K=3)
__global__ void knn_kernel(const float* __restrict__ dxyz, const float* __restrict__ sxyz,
                           int* __restrict__ idx, float* __restrict__ wgt) {
  __shared__ float sx[N2], sy[N2], sz[N2];
  int b = blockIdx.y;
  int tid = threadIdx.x;
  for (int j = tid; j < N2; j += 256) {
    sx[j] = sxyz[(b * 3 + 0) * N2 + j];
    sy[j] = sxyz[(b * 3 + 1) * N2 + j];
    sz[j] = sxyz[(b * 3 + 2) * N2 + j];
  }
  __syncthreads();
  int n = blockIdx.x * 256 + tid;
  float px = dxyz[(b * 3 + 0) * N1 + n];
  float py = dxyz[(b * 3 + 1) * N1 + n];
  float pz = dxyz[(b * 3 + 2) * N1 + n];
  float d0 = 3.4e38f, d1 = 3.4e38f, d2 = 3.4e38f;
  int i0 = 0, i1 = 0, i2 = 0;
  for (int j = 0; j < N2; ++j) {
    float dx = px - sx[j], dy = py - sy[j], dz = pz - sz[j];
    float d = dx * dx + dy * dy + dz * dz;
    if (d < d0)      { d2 = d1; i2 = i1; d1 = d0; i1 = i0; d0 = d; i0 = j; }
    else if (d < d1) { d2 = d1; i2 = i1; d1 = d;  i1 = j; }
    else if (d < d2) { d2 = d;  i2 = j; }
  }
  d0 = fmaxf(d0, 1e-10f); d1 = fmaxf(d1, 1e-10f); d2 = fmaxf(d2, 1e-10f);
  float w0 = 1.f / d0, w1 = 1.f / d1, w2 = 1.f / d2;
  float ws = w0 + w1 + w2;
  int p = b * N1 + n;
  idx[p * 3 + 0] = i0; idx[p * 3 + 1] = i1; idx[p * 3 + 2] = i2;
  wgt[p * 3 + 0] = w0 / ws; wgt[p * 3 + 1] = w1 / ws; wgt[p * 3 + 2] = w2 / ws;
}

// --------------------------------------- (B,C,N) -> (B,N,C) channel-contig
__global__ void transpose_bcn(const float* __restrict__ in, float* __restrict__ out,
                              int C, int N) {
  int id = blockIdx.x * 256 + threadIdx.x;
  int total = BB * C * N;
  if (id >= total) return;
  int n = id % N;
  int c = (id / N) % C;
  int b = id / (N * C);
  out[((size_t)b * N + n) * C + c] = in[id];
}

// ------------- pack weights into bf16 WMMA A-fragment layout (contiguous k = 16*(lane/16) + j)
__global__ void pack_w_kernel(const float* __restrict__ W, unsigned short* __restrict__ Apack,
                              int Cin) {
  int e = blockIdx.x * 256 + threadIdx.x;  // total = 256*Cin
  int j = e & 15;
  int lane = (e >> 4) & 31;
  int nkt = Cin >> 5;
  int kt = (e >> 9) % nkt;
  int mtile = e / (nkt << 9);
  int m = mtile * 16 + (lane & 15);
  int k = kt * 32 + j + ((lane >= 16) ? 16 : 0);
  Apack[e] = f2bf(W[m * Cin + k]);
}

// -------- build feat (col-major bf16, stride CIN1): [interp | g-interp | dense]
__global__ void build_feat_kernel(const float* __restrict__ sfT, const float* __restrict__ dfT,
                                  const int* __restrict__ idx, const float* __restrict__ wgt,
                                  unsigned short* __restrict__ featB) {
  int id = blockIdx.x * 256 + threadIdx.x;   // BB*N1*192 threads
  int c = id % 192;
  int p = id / 192;                           // b*N1 + n
  int b = p >> 12;                            // N1 = 4096
  size_t col0 = (size_t)p * 3;
  if (c < C2X) {
    int i0 = idx[p * 3 + 0], i1 = idx[p * 3 + 1], i2 = idx[p * 3 + 2];
    float w0 = wgt[p * 3 + 0], w1 = wgt[p * 3 + 1], w2 = wgt[p * 3 + 2];
    float g0 = sfT[((size_t)b * N2 + i0) * C2X + c];
    float g1 = sfT[((size_t)b * N2 + i1) * C2X + c];
    float g2 = sfT[((size_t)b * N2 + i2) * C2X + c];
    float it = w0 * g0 + w1 * g1 + w2 * g2;
    unsigned short itb = f2bf(it);
    featB[(col0 + 0) * CIN1 + c] = itb;
    featB[(col0 + 1) * CIN1 + c] = itb;
    featB[(col0 + 2) * CIN1 + c] = itb;
    featB[(col0 + 0) * CIN1 + C2X + c] = f2bf(g0 - it);
    featB[(col0 + 1) * CIN1 + C2X + c] = f2bf(g1 - it);
    featB[(col0 + 2) * CIN1 + C2X + c] = f2bf(g2 - it);
  } else {
    int cd = c - C2X;  // 0..63
    unsigned short v = f2bf(dfT[(size_t)p * C1X + cd]);
    featB[(col0 + 0) * CIN1 + 2 * C2X + cd] = v;
    featB[(col0 + 1) * CIN1 + 2 * C2X + cd] = v;
    featB[(col0 + 2) * CIN1 + 2 * C2X + cd] = v;
  }
}

// -------------------------------------------- bf16 WMMA GEMM + BN statistics
// Y[m, col] = sum_k W[m,k] * B[k,col];  M=256, Ncols=98304.
// Block: 8 waves = 4 m-tiles x 2 column halves -> 64(M) x 128(N) per block.
// The 32(K) x 128(N) B tile is double-buffered in LDS via async global->LDS
// copies (ASYNCcnt), shared by the 4 m-tile waves.  Fused per-channel
// sum/sum^2 reduction feeds training-mode BatchNorm.
__global__ __launch_bounds__(256) void wmma_gemm_kernel(
    const unsigned short* __restrict__ Apack,
    const unsigned short* __restrict__ Bcm,
    unsigned short* __restrict__ Ycm,
    float* __restrict__ stats, int Cin) {
  __shared__ __attribute__((aligned(32))) unsigned short Bs[2][128 * 32]; // 2 x 8KB
  __shared__ float lsum[64];
  __shared__ float lsq[64];

  int tid = threadIdx.x;
  if (tid < 64) { lsum[tid] = 0.f; lsq[tid] = 0.f; }

  int lane = tid & 31;
  int wv = tid >> 5;
  int mloc4 = wv & 3;            // m-tile within block
  int colhalf = wv >> 2;         // 0/1
  int mtile = blockIdx.y * 4 + mloc4;
  int col0 = blockIdx.x * 128;
  int nkt = Cin >> 5;
  int halfsel = (lane >= 16) ? 16 : 0;

  // staging: 512 chunks of 16B per k-step; each thread moves chunks tid, tid+256
  int sc0 = tid >> 2,         sq0 = tid & 3;
  int sc1 = (tid + 256) >> 2, sq1 = (tid + 256) & 3;
  const unsigned short* g0 = Bcm + (size_t)(col0 + sc0) * Cin + sq0 * 8;
  const unsigned short* g1 = Bcm + (size_t)(col0 + sc1) * Cin + sq1 * 8;
  unsigned short* l0a = &Bs[0][sc0 * 32 + sq0 * 8];
  unsigned short* l0b = &Bs[0][sc1 * 32 + sq1 * 8];
  unsigned short* l1a = &Bs[1][sc0 * 32 + sq0 * 8];
  unsigned short* l1b = &Bs[1][sc1 * 32 + sq1 * 8];

  const v16bf* Ap = (const v16bf*)Apack + (size_t)mtile * nkt * 32 + lane;
  int cl = colhalf * 64 + (lane & 15);   // LDS column for subtile 0

  v8f acc0 = {}, acc1 = {}, acc2 = {}, acc3 = {};

  // prime buffer 0
  stage16(g0, l0a);
  stage16(g1, l0b);

  for (int kt = 0; kt < nkt; ++kt) {
    wait_async_copies();
    __syncthreads();                     // current buffer visible to all waves
    int cur = kt & 1;
    if (kt + 1 < nkt) {                  // overlap next-tile copy with WMMAs
      int ko = (kt + 1) * 32;
      stage16(g0 + ko, cur ? l0a : l1a);
      stage16(g1 + ko, cur ? l0b : l1b);
    }
    v16bf a = Ap[kt * 32];
    const unsigned short* bs = &Bs[cur][0];
    v16bf f0 = *(const v16bf*)(bs + (cl + 0)  * 32 + halfsel);
    v16bf f1 = *(const v16bf*)(bs + (cl + 16) * 32 + halfsel);
    v16bf f2 = *(const v16bf*)(bs + (cl + 32) * 32 + halfsel);
    v16bf f3 = *(const v16bf*)(bs + (cl + 48) * 32 + halfsel);
    acc0 = __builtin_amdgcn_wmma_f32_16x16x32_bf16(false, a, false, f0, (short)0, acc0, false, false);
    acc1 = __builtin_amdgcn_wmma_f32_16x16x32_bf16(false, a, false, f1, (short)0, acc1, false, false);
    acc2 = __builtin_amdgcn_wmma_f32_16x16x32_bf16(false, a, false, f2, (short)0, acc2, false, false);
    acc3 = __builtin_amdgcn_wmma_f32_16x16x32_bf16(false, a, false, f3, (short)0, acc3, false, false);
    __syncthreads();                     // reads done before buffer is re-filled
  }

  // C/D layout: VGPR r -> m = mtile*16 + r + (lane>=16 ? 8 : 0); n = lane&15
  int mhalf = (lane >= 16) ? 8 : 0;
  int mrow = mtile * 16 + mhalf;
  float psum[8], psq[8];
  for (int r = 0; r < 8; ++r) {
    psum[r] = acc0[r] + acc1[r] + acc2[r] + acc3[r];
    psq[r]  = acc0[r] * acc0[r] + acc1[r] * acc1[r] + acc2[r] * acc2[r] + acc3[r] * acc3[r];
  }

  // store Y col-major bf16 (8 consecutive m per lane -> one 16B store each)
  {
    unsigned short o[8];
    int colg = col0 + colhalf * 64 + (lane & 15);
    for (int r = 0; r < 8; ++r) o[r] = f2bf(acc0[r]);
    *(uint4*)(Ycm + (size_t)(colg + 0)  * CH + mrow) = *(uint4*)o;
    for (int r = 0; r < 8; ++r) o[r] = f2bf(acc1[r]);
    *(uint4*)(Ycm + (size_t)(colg + 16) * CH + mrow) = *(uint4*)o;
    for (int r = 0; r < 8; ++r) o[r] = f2bf(acc2[r]);
    *(uint4*)(Ycm + (size_t)(colg + 32) * CH + mrow) = *(uint4*)o;
    for (int r = 0; r < 8; ++r) o[r] = f2bf(acc3[r]);
    *(uint4*)(Ycm + (size_t)(colg + 48) * CH + mrow) = *(uint4*)o;
  }

  int li = mloc4 * 16 + mhalf;
  for (int r = 0; r < 8; ++r) {
    atomicAdd(&lsum[li + r], psum[r]);
    atomicAdd(&lsq[li + r], psq[r]);
  }
  __syncthreads();
  if (tid < 64) {
    atomicAdd(&stats[blockIdx.y * 64 + tid], lsum[tid]);
    atomicAdd(&stats[CH + blockIdx.y * 64 + tid], lsq[tid]);
  }
}

// ---------------------------------------------------------- BN finalization
__global__ void finalize_bn_kernel(const float* __restrict__ stats,
                                   const float* __restrict__ gamma,
                                   const float* __restrict__ beta,
                                   float* __restrict__ ab) {
  int m = threadIdx.x;
  float cnt = (float)NCOLS;
  float mean = stats[m] / cnt;
  float var = stats[CH + m] / cnt - mean * mean;
  float inv = rsqrtf(var + BNEPS);
  float a = gamma[m] * inv;
  ab[m] = a;
  ab[CH + m] = beta[m] - mean * a;
}

// ---------------------------------------------------- elementwise BN + ReLU
__global__ void bn_relu_kernel(const unsigned short* __restrict__ Y,
                               const float* __restrict__ ab,
                               unsigned short* __restrict__ H) {
  int id = blockIdx.x * 256 + threadIdx.x;  // NCOLS*CH/8 threads
  size_t base = (size_t)id * 8;
  int m0 = (int)(base & (CH - 1));
  uint4 in = *(const uint4*)(Y + base);
  unsigned short* ip = (unsigned short*)&in;
  unsigned short o[8];
  for (int r = 0; r < 8; ++r) {
    int m = m0 + r;
    float v = ab[m] * bf2f(ip[r]) + ab[CH + m];
    o[r] = f2bf(fmaxf(v, 0.f));
  }
  *(uint4*)(H + base) = *(uint4*)o;
}

// --------------------------------------- BN2 + ReLU + mean over K, f32 out
__global__ void final_kernel(const unsigned short* __restrict__ Y2,
                             const float* __restrict__ ab,
                             float* __restrict__ out) {
  int id = blockIdx.x * 256 + threadIdx.x;  // BB*N1*CH, m inner
  int m = id & (CH - 1);
  int p = id >> 8;                           // b*N1 + n
  float a = ab[m], bb = ab[CH + m];
  size_t cb = (size_t)p * 3 * CH + m;
  float v0 = fmaxf(a * bf2f(Y2[cb]) + bb, 0.f);
  float v1 = fmaxf(a * bf2f(Y2[cb + CH]) + bb, 0.f);
  float v2 = fmaxf(a * bf2f(Y2[cb + 2 * CH]) + bb, 0.f);
  int b = p >> 12;
  int n = p & (N1 - 1);
  out[((size_t)b * CH + m) * N1 + n] = (v0 + v1 + v2) * (1.f / 3.f);
}

__global__ void zero_stats_kernel(float* __restrict__ s1, float* __restrict__ s2) {
  int t = threadIdx.x;  // 512
  s1[t] = 0.f;
  s2[t] = 0.f;
}

extern "C" void kernel_launch(void* const* d_in, const int* in_sizes, int n_in,
                              void* d_out, int out_size, void* d_ws, size_t ws_size,
                              hipStream_t stream) {
  const float* dxyz = (const float*)d_in[0];
  const float* sxyz = (const float*)d_in[1];
  const float* df   = (const float*)d_in[2];
  const float* sf   = (const float*)d_in[3];
  const float* W1   = (const float*)d_in[4];
  const float* g1   = (const float*)d_in[5];
  const float* be1  = (const float*)d_in[6];
  const float* W2   = (const float*)d_in[7];
  const float* g2   = (const float*)d_in[8];
  const float* be2  = (const float*)d_in[9];
  float* out = (float*)d_out;

  char* ws = (char*)d_ws;
  size_t off = 0;
  int*   idx    = (int*)(ws + off);            off += (size_t)NCOLS * 4;
  float* wgt    = (float*)(ws + off);          off += (size_t)NCOLS * 4;
  unsigned short* Apack1 = (unsigned short*)(ws + off); off += (size_t)CH * CIN1 * 2;
  unsigned short* Apack2 = (unsigned short*)(ws + off); off += (size_t)CH * CH * 2;
  float* stats1 = (float*)(ws + off);          off += 2048;
  float* stats2 = (float*)(ws + off);          off += 2048;
  float* ab1    = (float*)(ws + off);          off += 2048;
  float* ab2    = (float*)(ws + off);          off += 2048;
  float* sfT    = (float*)(ws + off);          off += (size_t)BB * N2 * C2X * 4;   // 4 MB
  float* dfT    = (float*)(ws + off);          off += (size_t)BB * N1 * C1X * 4;   // 8 MB
  unsigned short* R1 = (unsigned short*)(ws + off); off += (size_t)NCOLS * CIN1 * 2; // 63 MB
  unsigned short* R2 = (unsigned short*)(ws + off); off += (size_t)NCOLS * CH * 2;   // 50 MB
  unsigned short* featB = R1;
  unsigned short* h     = R1;   // reuses featB region after gemm1 consumed it
  unsigned short* y1    = R2;
  unsigned short* y2    = R2;   // reuses y1 region after bn_relu consumed it

  zero_stats_kernel<<<1, 512, 0, stream>>>(stats1, stats2);
  knn_kernel<<<dim3(N1 / 256, BB), 256, 0, stream>>>(dxyz, sxyz, idx, wgt);
  transpose_bcn<<<(BB * C2X * N2) / 256, 256, 0, stream>>>(sf, sfT, C2X, N2);
  transpose_bcn<<<(BB * C1X * N1) / 256, 256, 0, stream>>>(df, dfT, C1X, N1);
  pack_w_kernel<<<(CH * CIN1) / 256, 256, 0, stream>>>(W1, Apack1, CIN1);
  pack_w_kernel<<<(CH * CH) / 256, 256, 0, stream>>>(W2, Apack2, CH);
  build_feat_kernel<<<(BB * N1 * 192) / 256, 256, 0, stream>>>(sfT, dfT, idx, wgt, featB);
  wmma_gemm_kernel<<<dim3(NCOLS / 128, 4), 256, 0, stream>>>(Apack1, featB, y1, stats1, CIN1);
  finalize_bn_kernel<<<1, 256, 0, stream>>>(stats1, g1, be1, ab1);
  bn_relu_kernel<<<((size_t)NCOLS * CH / 8) / 256, 256, 0, stream>>>(y1, ab1, h);
  wmma_gemm_kernel<<<dim3(NCOLS / 128, 4), 256, 0, stream>>>(Apack2, h, y2, stats2, CH);
  finalize_bn_kernel<<<1, 256, 0, stream>>>(stats2, g2, be2, ab2);
  final_kernel<<<(BB * N1 * CH) / 256, 256, 0, stream>>>(y2, ab2, out);
}
